// GraphConvolutionalNetwork_18451179503794
// MI455X (gfx1250) — matrix operand order repaired and checked by reading, not compiled, not run on hardware
//
#include <hip/hip_runtime.h>

// ---------------------------------------------------------------------------
// GCN: 3 layers (128->256->256->64) + graph sum pooling on MI455X (gfx1250).
// GEMMs use v_wmma_f32_16x16x32_bf16 (fp32 accumulate); edge scatter uses
// fp32 global atomics (AGG buffers are L2-resident: 51MB << 192MB L2).
// ---------------------------------------------------------------------------

typedef __attribute__((ext_vector_type(16))) __bf16 v16bf;
typedef __attribute__((ext_vector_type(8)))  __bf16 v8bf;
typedef __attribute__((ext_vector_type(8)))  float  v8f;

// ------------------------------ utility kernels -----------------------------

__global__ void zero_f32(float* __restrict__ p, int n) {
    int i = blockIdx.x * 256 + threadIdx.x;
    if (i < n) p[i] = 0.0f;
}

__global__ void f32_to_bf16_k(const float* __restrict__ s, __bf16* __restrict__ d, int n) {
    int i = blockIdx.x * 256 + threadIdx.x;
    if (i < n) d[i] = (__bf16)s[i];
}

__global__ void degree_kernel(const int* __restrict__ snd, const int* __restrict__ rcv,
                              float* __restrict__ deg_s, float* __restrict__ deg_r, int E) {
    int e = blockIdx.x * 256 + threadIdx.x;
    if (e < E) {
        atomicAdd(&deg_s[snd[e]], 1.0f);
        atomicAdd(&deg_r[rcv[e]], 1.0f);
    }
}

__global__ void rsqrt_inplace(float* __restrict__ ds, float* __restrict__ dr, int n) {
    int i = blockIdx.x * 256 + threadIdx.x;
    if (i < n) {
        ds[i] = rsqrtf(fmaxf(ds[i], 1.0f));
        dr[i] = rsqrtf(fmaxf(dr[i], 1.0f));
    }
}

// One wave (32 lanes) per edge: coalesced row gather + f32 atomic scatter.
__global__ void edge_aggregate(const float* __restrict__ H, const int* __restrict__ snd,
                               const int* __restrict__ rcv, float* __restrict__ AGG,
                               int E, int D) {
    int t = blockIdx.x * 256 + threadIdx.x;
    int e = t >> 5;
    int lane = t & 31;
    if (e >= E) return;
    int s = snd[e];
    int r = rcv[e];
    const float* src = H + (size_t)s * D;
    float* dst = AGG + (size_t)r * D;
    for (int c = lane; c < D; c += 32)
        atomicAdd(&dst[c], src[c]);
}

// AGG * inv_r[node], ReLU, convert to bf16 for next layer's A matrix.
__global__ void post_relu_bf16(const float* __restrict__ AGG, const float* __restrict__ invr,
                               __bf16* __restrict__ dst, int total, int shift) {
    int i = blockIdx.x * 256 + threadIdx.x;
    if (i < total) {
        int nn = i >> shift;
        float v = AGG[i] * invr[nn];
        v = fmaxf(v, 0.0f);
        dst[i] = (__bf16)v;
    }
}

// Final: AGG3 * inv_r -> atomic add into out[batch[n]][c]  (D = 64)
__global__ void pool_kernel(const float* __restrict__ AGG, const float* __restrict__ invr,
                            const int* __restrict__ batch, float* __restrict__ out, int total) {
    int i = blockIdx.x * 256 + threadIdx.x;
    if (i < total) {
        int nn = i >> 6;
        int c = i & 63;
        atomicAdd(&out[batch[nn] * 64 + c], AGG[i] * invr[nn]);
    }
}

// ------------------------------ WMMA GEMM -----------------------------------
// Out[M,N] = (A[M,K](bf16) @ W[K,N](bf16) + bias[N]) * rowScale[row]
// Block tile 64x64, 8 waves; wave w: rows (w&3)*16..+15, cols (w>>2)*32..+31
// (two 16x16x32 bf16 WMMAs per K-step). K multiple of 32, N multiple of 64.
__global__ __launch_bounds__(256) void gemm_bf16_wmma(
    const __bf16* __restrict__ A, const __bf16* __restrict__ W,
    const float* __restrict__ bias, const float* __restrict__ rowScale,
    float* __restrict__ Out, int M, int N, int K) {
    // Padded stride 40 halfwords (80B) keeps ds_load_b128 16B-aligned and
    // spreads the 16 row-readers across banks.
    __shared__ __bf16 As[64 * 40];   // [row][k]   row-major, 32 valid k
    __shared__ __bf16 Bs[64 * 40];   // [n][k]     transposed, 32 valid k

    const int tid  = threadIdx.x;
    const int lane = tid & 31;
    const int wid  = tid >> 5;
    const int rowStrip = (wid & 3) << 4;   // 0,16,32,48
    const int colHalf  = (wid >> 2) << 5;  // 0,32
    const int blockRow = blockIdx.y << 6;
    const int blockCol = blockIdx.x << 6;

    v8f acc0 = {};
    v8f acc1 = {};

    for (int k0 = 0; k0 < K; k0 += 32) {
        // --- A tile 64x32: one b128 global load per thread ---
        {
            int r  = tid >> 2;
            int c8 = (tid & 3) << 3;
            int gr = blockRow + r;
            v8bf av = {};
            if (gr < M) av = *(const v8bf*)(A + (size_t)gr * K + k0 + c8);
            *(v8bf*)(&As[r * 40 + c8]) = av;
        }
        // --- B tile 32x64, transposed into [n][k] ---
        {
            int k  = tid >> 3;             // 0..31
            int c8 = (tid & 7) << 3;       // n chunk of 8
            v8bf wv = *(const v8bf*)(W + (size_t)(k0 + k) * N + blockCol + c8);
#pragma unroll
            for (int j = 0; j < 8; ++j) Bs[(c8 + j) * 40 + k] = wv[j];
        }
        __syncthreads();

        // A fragment: lanes 0-15 hold K {0..7,16..23}, lanes 16-31 {8..15,24..31}
        const int ar  = rowStrip + (lane & 15);
        const int koA = (lane < 16) ? 0 : 8;
        union { v16bf v; v8bf h[2]; } af;
        af.h[0] = *(const v8bf*)(&As[ar * 40 + koA]);
        af.h[1] = *(const v8bf*)(&As[ar * 40 + 16 + koA]);

        // B fragments: lane n = base + (lane&15); K pairs contiguous in [n][k]
        const int koB = (lane < 16) ? 0 : 8;
        const int bn0 = colHalf + (lane & 15);
        union { v16bf v; v8bf h[2]; } bf0, bf1;
        bf0.h[0] = *(const v8bf*)(&Bs[bn0 * 40 + koB]);
        bf0.h[1] = *(const v8bf*)(&Bs[bn0 * 40 + koB + 16]);
        const int bn1 = bn0 + 16;
        bf1.h[0] = *(const v8bf*)(&Bs[bn1 * 40 + koB]);
        bf1.h[1] = *(const v8bf*)(&Bs[bn1 * 40 + koB + 16]);

        acc0 = __builtin_amdgcn_wmma_f32_16x16x32_bf16(
            false, af.v, false, bf0.v, (short)0, acc0, false, false);
        acc1 = __builtin_amdgcn_wmma_f32_16x16x32_bf16(
            false, af.v, false, bf1.v, (short)0, acc1, false, false);
        __syncthreads();
    }

    // C/D layout: VGPR j, lanes 0-15 -> M=j, lanes 16-31 -> M=8+j; N = lane&15
    const int rbase = blockRow + rowStrip + ((lane >> 4) << 3);
    const int cA = blockCol + colHalf + (lane & 15);
    const int cB = cA + 16;
    const float bA = bias[cA];
    const float bB = bias[cB];
#pragma unroll
    for (int j = 0; j < 8; ++j) {
        int r = rbase + j;
        if (r < M) {
            float sc = rowScale[r];
            Out[(size_t)r * N + cA] = (acc0[j] + bA) * sc;
            Out[(size_t)r * N + cB] = (acc1[j] + bB) * sc;
        }
    }
}

// ------------------------------ launch --------------------------------------

extern "C" void kernel_launch(void* const* d_in, const int* in_sizes, int n_in,
                              void* d_out, int out_size, void* d_ws, size_t ws_size,
                              hipStream_t stream) {
    const float* x  = (const float*)d_in[0];
    const float* W1 = (const float*)d_in[1];
    const float* b1 = (const float*)d_in[2];
    const float* W2 = (const float*)d_in[3];
    const float* b2 = (const float*)d_in[4];
    const float* W3 = (const float*)d_in[5];
    const float* b3 = (const float*)d_in[6];
    const int* snd   = (const int*)d_in[7];
    const int* rcv   = (const int*)d_in[8];
    const int* batch = (const int*)d_in[9];
    float* out = (float*)d_out;

    const int IN_DIM = 128, HID = 256, OUT_DIM = 64;
    const int n_nodes = in_sizes[0] / IN_DIM;
    const int n_edges = in_sizes[7];

    // workspace carve-out (256B aligned)
    char* wsp = (char*)d_ws;
    auto alloc = [&](size_t bytes) -> char* {
        char* p = wsp;
        wsp += (bytes + 255) & ~(size_t)255;
        return p;
    };
    float*  inv_s = (float*)alloc((size_t)n_nodes * 4);
    float*  inv_r = (float*)alloc((size_t)n_nodes * 4);
    __bf16* Abf   = (__bf16*)alloc((size_t)n_nodes * HID * 2);
    float*  H     = (float*)alloc((size_t)n_nodes * HID * 4);
    float*  AGG   = (float*)alloc((size_t)n_nodes * HID * 4);
    __bf16* W1b   = (__bf16*)alloc((size_t)IN_DIM * HID * 2);
    __bf16* W2b   = (__bf16*)alloc((size_t)HID * HID * 2);
    __bf16* W3b   = (__bf16*)alloc((size_t)HID * OUT_DIM * 2);

    dim3 blk(256);
    auto cdiv = [](int a, int b) { return (a + b - 1) / b; };

    // ---- degrees -> rsqrt scales ----
    zero_f32<<<cdiv(n_nodes, 256), blk, 0, stream>>>(inv_s, n_nodes);
    zero_f32<<<cdiv(n_nodes, 256), blk, 0, stream>>>(inv_r, n_nodes);
    degree_kernel<<<cdiv(n_edges, 256), blk, 0, stream>>>(snd, rcv, inv_s, inv_r, n_edges);
    rsqrt_inplace<<<cdiv(n_nodes, 256), blk, 0, stream>>>(inv_s, inv_r, n_nodes);

    // ---- bf16 conversions ----
    f32_to_bf16_k<<<cdiv(n_nodes * IN_DIM, 256), blk, 0, stream>>>(x, Abf, n_nodes * IN_DIM);
    f32_to_bf16_k<<<cdiv(IN_DIM * HID, 256), blk, 0, stream>>>(W1, W1b, IN_DIM * HID);
    f32_to_bf16_k<<<cdiv(HID * HID, 256), blk, 0, stream>>>(W2, W2b, HID * HID);
    f32_to_bf16_k<<<cdiv(HID * OUT_DIM, 256), blk, 0, stream>>>(W3, W3b, HID * OUT_DIM);

    dim3 gWide(HID / 64, cdiv(n_nodes, 64));    // N=256
    dim3 gNarrow(OUT_DIM / 64, cdiv(n_nodes, 64)); // N=64

    // ---- layer 1: 128 -> 256 ----
    gemm_bf16_wmma<<<gWide, blk, 0, stream>>>(Abf, W1b, b1, inv_s, H, n_nodes, HID, IN_DIM);
    zero_f32<<<cdiv(n_nodes * HID, 256), blk, 0, stream>>>(AGG, n_nodes * HID);
    edge_aggregate<<<cdiv(n_edges * 32, 256), blk, 0, stream>>>(H, snd, rcv, AGG, n_edges, HID);
    post_relu_bf16<<<cdiv(n_nodes * HID, 256), blk, 0, stream>>>(AGG, inv_r, Abf, n_nodes * HID, 8);

    // ---- layer 2: 256 -> 256 ----
    gemm_bf16_wmma<<<gWide, blk, 0, stream>>>(Abf, W2b, b2, inv_s, H, n_nodes, HID, HID);
    zero_f32<<<cdiv(n_nodes * HID, 256), blk, 0, stream>>>(AGG, n_nodes * HID);
    edge_aggregate<<<cdiv(n_edges * 32, 256), blk, 0, stream>>>(H, snd, rcv, AGG, n_edges, HID);
    post_relu_bf16<<<cdiv(n_nodes * HID, 256), blk, 0, stream>>>(AGG, inv_r, Abf, n_nodes * HID, 8);

    // ---- layer 3: 256 -> 64 (no relu), then graph pooling ----
    gemm_bf16_wmma<<<gNarrow, blk, 0, stream>>>(Abf, W3b, b3, inv_s, H, n_nodes, OUT_DIM, HID);
    zero_f32<<<cdiv(n_nodes * OUT_DIM, 256), blk, 0, stream>>>(AGG, n_nodes * OUT_DIM);
    edge_aggregate<<<cdiv(n_edges * 32, 256), blk, 0, stream>>>(H, snd, rcv, AGG, n_edges, OUT_DIM);
    zero_f32<<<cdiv(out_size, 256), blk, 0, stream>>>(out, out_size);
    pool_kernel<<<cdiv(n_nodes * OUT_DIM, 256), blk, 0, stream>>>(AGG, inv_r, batch, out,
                                                                  n_nodes * OUT_DIM);
}
